// DEC_59184649339515
// MI455X (gfx1250) — matrix operand order repaired and checked by reading, compiled, not run on hardware
//
#include <hip/hip_runtime.h>

typedef __attribute__((ext_vector_type(16))) __bf16 bf16x16;
typedef __attribute__((ext_vector_type(8)))  __bf16 bf16x8;
typedef __attribute__((ext_vector_type(8)))  float  v8f;

constexpr int kB = 1024;
constexpr int kL = 64;
constexpr int kH = 512;
constexpr int kD = 10;
constexpr int k3H = 1536;
constexpr int kTileElems = 16 * kH;   // elements in one 16 x 512 fragment buffer
constexpr int kNTStride = 32 * 16;    // elements per B-fragment N-tile block (512)
constexpr int kKIters = kH / 32;      // 16 K-steps per 512-deep GEMM

// sched_group_barrier masks
#define SG_WMMA 0x008
#define SG_VMEM_RD 0x020
#define SG_DS_RD 0x100

__device__ __forceinline__ float fast_rcp(float x) { return __builtin_amdgcn_rcpf(x); }
__device__ __forceinline__ float fast_exp(float x) { return __builtin_amdgcn_exp2f(x * 1.44269504f); }
__device__ __forceinline__ float fast_tanh(float x) {
  float e = __builtin_amdgcn_exp2f(x * 2.885390082f);   // exp(2x)
  return 1.0f - 2.0f * fast_rcp(e + 1.0f);
}
__device__ __forceinline__ float fast_sigmoid(float x) {
  return fast_rcp(1.0f + __builtin_amdgcn_exp2f(x * -1.44269504f));
}

// Index of element (row m, K-col e) inside a 16 x K bf16 A-fragment buffer.
// Per ISA 7.12.2: lane = half*16 + m ; VGPR-half t holds K = 8*half + t + (t>=8 ? 8 : 0).
__device__ __forceinline__ int afrag_idx(int m, int e) {
  int kt = e >> 5;
  int r5 = e & 31;
  int half = (r5 >> 3) & 1;
  int rp = r5 - (half << 3);          // {0..7} U {16..23}
  int t = (rp < 8) ? rp : (rp - 8);
  return (((kt << 5) + (half << 4) + m) << 4) + t;
}

#define WMMA_BF16(A, Bv, C) \
  __builtin_amdgcn_wmma_f32_16x16x32_bf16(false, (A), false, (Bv), (short)0, (C), false, false)

// Four adjacent 16x16 output tiles sharing one A stream; NT (width of B) fixed at 32.
// A and B loads are software-pipelined one K-step ahead; sched_group_barriers pin the
// [DS(next-A), VMEM(next-B), WMMA(cur)] steady-state so waits stay partial.
__device__ __forceinline__ void wmma_rows4(const __bf16* af, const __bf16* bf,
                                           int nt0, int lane,
                                           v8f& c0, v8f& c1, v8f& c2, v8f& c3) {
  const int laneE = lane << 4;
  const __bf16* bb = bf + ((size_t)nt0 << 9) + laneE;
  constexpr int BSTEP = 32 * 512;               // elements per kt step (NT=32)
  bf16x16 a = *(const bf16x16*)(af + laneE);
  bf16x16 b0 = *(const bf16x16*)(bb);
  bf16x16 b1 = *(const bf16x16*)(bb + kNTStride);
  bf16x16 b2 = *(const bf16x16*)(bb + 2 * kNTStride);
  bf16x16 b3 = *(const bf16x16*)(bb + 3 * kNTStride);
  __builtin_amdgcn_sched_group_barrier(SG_DS_RD, 2, 0);    // absorbs A(kt=0)
  __builtin_amdgcn_sched_group_barrier(SG_VMEM_RD, 8, 0);  // absorbs B(kt=0)
#pragma unroll
  for (int kt = 0; kt < kKIters; ++kt) {
    bf16x16 an, n0, n1, n2, n3;
    if (kt < kKIters - 1) {
      const __bf16* nb = bb + (size_t)(kt + 1) * BSTEP;
      an = *(const bf16x16*)(af + (kt + 1) * 512 + laneE);
      n0 = *(const bf16x16*)(nb);
      n1 = *(const bf16x16*)(nb + kNTStride);
      n2 = *(const bf16x16*)(nb + 2 * kNTStride);
      n3 = *(const bf16x16*)(nb + 3 * kNTStride);
    }
    c0 = WMMA_BF16(a, b0, c0);
    c1 = WMMA_BF16(a, b1, c1);
    c2 = WMMA_BF16(a, b2, c2);
    c3 = WMMA_BF16(a, b3, c3);
    __builtin_amdgcn_sched_group_barrier(SG_DS_RD, 2, 0);
    __builtin_amdgcn_sched_group_barrier(SG_VMEM_RD, 8, 0);
    __builtin_amdgcn_sched_group_barrier(SG_WMMA, 4, 0);
    if (kt < kKIters - 1) { a = an; b0 = n0; b1 = n1; b2 = n2; b3 = n3; }
  }
}

// Pack a (N x K) fp32 weight (B[k][n] = src[n*stride + k + colOff]) into bf16 WMMA
// B-fragment order: dst[((kt*NT+nt)*32 + lane)*16 + t], lane = half*16+n, K = kt*32+half*8+(t&7)+(t>>3)*16.
__global__ void pack_bfrag_kernel(const float* __restrict__ src, __bf16* __restrict__ dst,
                                  int N, int srcStride, int colOff) {
  int fid = blockIdx.x * blockDim.x + threadIdx.x;
  int total = N << 9;                 // N * 512
  if (fid >= total) return;
  int t = fid & 15;
  int lane = (fid >> 4) & 31;
  int tile = fid >> 9;
  int NT = N >> 4;
  int nt = tile % NT;
  int kt = tile / NT;
  int half = lane >> 4;
  int n = (nt << 4) + (lane & 15);
  int k = (kt << 5) + (half << 3) + (t & 7) + ((t >> 3) << 4);
  dst[fid] = (__bf16)src[(size_t)n * srcStride + k + colOff];
}

// enc = tanh(received @ fc_w.T + fc_b), bf16, rows flattened (b*L + l).
__global__ void encode_kernel(const float* __restrict__ received,
                              const float* __restrict__ fc_w,
                              const float* __restrict__ fc_b,
                              __bf16* __restrict__ enc_bf) {
  size_t idx = (size_t)blockIdx.x * blockDim.x + threadIdx.x;
  int h = (int)(idx & (kH - 1));
  size_t bl = idx >> 9;
  const float* r = received + bl * 3;
  float e = fast_tanh(r[0] * fc_w[h * 3 + 0] + r[1] * fc_w[h * 3 + 1] +
                      r[2] * fc_w[h * 3 + 2] + fc_b[h]);
  enc_bf[idx] = (__bf16)e;
}

// encproj = enc @ attn_w[:, H:].T  (step-invariant half of the attention energy)
__global__ void __launch_bounds__(256)
encproj_kernel(const __bf16* __restrict__ enc_bf, const __bf16* __restrict__ w2t,
               __bf16* __restrict__ encproj) {
  __shared__ __bf16 af[kTileElems];
  const int tid = threadIdx.x, lane = tid & 31, wv = tid >> 5;
  const size_t r0 = (size_t)blockIdx.x * 16;
  for (int lin = tid; lin < kTileElems; lin += 256) {
    int m = lin >> 9, col = lin & (kH - 1);
    af[afrag_idx(m, col)] = enc_bf[(r0 + m) * kH + col];
  }
  __syncthreads();
  const int half = lane >> 4, n = lane & 15;
  int nt0 = wv * 4;
  v8f c0 = {}, c1 = {}, c2 = {}, c3 = {};
  wmma_rows4(af, w2t, nt0, lane, c0, c1, c2, c3);
#pragma unroll
  for (int i = 0; i < 8; ++i) {
    size_t row = r0 + half * 8 + i;
    encproj[row * kH + (nt0 + 0) * 16 + n] = (__bf16)c0[i];
    encproj[row * kH + (nt0 + 1) * 16 + n] = (__bf16)c1[i];
    encproj[row * kH + (nt0 + 2) * 16 + n] = (__bf16)c2[i];
    encproj[row * kH + (nt0 + 3) * 16 + n] = (__bf16)c3[i];
  }
}

// Persistent recurrent kernel: one 16-row batch tile per WG, 64 steps.
// h states kept in LDS as bf16 WMMA A-fragments, double-buffered (cur/nxt) so the
// GRU gate GEMMs, activations and state update fuse into one register-resident pass.
__global__ void __launch_bounds__(256, 1)
decoder_kernel(const __bf16* __restrict__ enc_bf,
               const __bf16* __restrict__ encproj,
               const __bf16* __restrict__ w1t,
               const __bf16* __restrict__ wgru,
               const float* __restrict__ g1bih, const float* __restrict__ g1bhh,
               const float* __restrict__ g2bih, const float* __restrict__ g2bhh,
               const float* __restrict__ v_w, const float* __restrict__ out_w,
               float* __restrict__ d1, float* __restrict__ d2) {
  extern __shared__ char smem[];
  __bf16* hfrag = (__bf16*)smem;                        // [2 buf][4 states][8192]
  __bf16* cfrag = hfrag + 8 * kTileElems;               // [2 dec][8192]
  float* sc1 = (float*)(cfrag + 2 * kTileElems);        // sproj dec0
  float* sc2 = sc1 + kTileElems;                        // sproj dec1
  float* logit = sc2 + kTileElems;                      // 2 x 16 x 64
  float* red2  = logit + 2 * 16 * kL;                   // [8 waves][16 rows]
  float* vws   = red2 + 128;                            // v_w (512)

  const int tid = threadIdx.x;
  const int lane = tid & 31;
  const int wv = tid >> 5;
  const int half = lane >> 4;
  const int nn16 = lane & 15;
  const int b0 = blockIdx.x * 16;

  // h0 = enc[:, L-1, :] broadcast to all 4 GRU states (buffer 0)
  for (int lin = tid; lin < kTileElems; lin += 256) {
    int m = lin >> 9, col = lin & (kH - 1);
    __bf16 v = enc_bf[((size_t)(b0 + m) * kL + (kL - 1)) * kH + col];
    int fi = afrag_idx(m, col);
    hfrag[fi] = v;
    hfrag[kTileElems + fi] = v;
    hfrag[2 * kTileElems + fi] = v;
    hfrag[3 * kTileElems + fi] = v;
  }
  for (int i = tid; i < kH; i += 256) vws[i] = v_w[i];
  __syncthreads();

  int cur = 0;
  for (int step = 0; step < kL; ++step) {
    const int nxt = cur ^ 1;
    // ---- P1: s-projection  h_top @ W1^T  (both decoders) ----
    {
      int dec = wv >> 2;                       // waves 0-3 -> dec0, 4-7 -> dec1
      const __bf16* af = hfrag + (size_t)(cur * 4 + dec * 2 + 1) * kTileElems;
      float* dst = dec ? sc2 : sc1;
#pragma unroll 1
      for (int pass = 0; pass < 2; ++pass) {
        int nt0 = (wv & 3) * 8 + pass * 4;
        v8f c0 = {}, c1 = {}, c2 = {}, c3 = {};
        wmma_rows4(af, w1t, nt0, lane, c0, c1, c2, c3);
#pragma unroll
        for (int i = 0; i < 8; ++i) {
          int row = (half * 8 + i) * kH;
          dst[row + (nt0 + 0) * 16 + nn16] = c0[i];
          dst[row + (nt0 + 1) * 16 + nn16] = c1[i];
          dst[row + (nt0 + 2) * 16 + nn16] = c2[i];
          dst[row + (nt0 + 3) * 16 + nn16] = c3[i];
        }
      }
    }
    __syncthreads();
    // ---- P2a: attention logits  sum_h v[h]*tanh(sproj + encproj) ----
    for (int q = 0; q < 8; ++q) {
      int jj = q * 256 + tid;                  // 2048 (dec,b,l) jobs
      int dec = jj >> 10;
      int b = (jj >> 6) & 15;
      int l = jj & 63;
      const float* sp = (dec ? sc2 : sc1) + b * kH;
      const __bf16* ep = encproj + ((size_t)(b0 + b) * kL + l) * kH;
      float pacc[4] = {};
      for (int h16 = 0; h16 < kH; h16 += 16) {
        bf16x16 ev = *(const bf16x16*)(ep + h16);
#pragma unroll
        for (int u = 0; u < 16; ++u)
          pacc[u & 3] += vws[h16 + u] * fast_tanh(sp[h16 + u] + (float)ev[u]);
      }
      logit[(dec * 16 + b) * kL + l] = (pacc[0] + pacc[1]) + (pacc[2] + pacc[3]);
    }
    __syncthreads();
    // ---- P2b: softmax over L (32 rows) ----
    if (tid < 32) {
      float* row = logit + tid * kL;
      float mx = row[0];
      for (int l = 1; l < kL; ++l) mx = fmaxf(mx, row[l]);
      float s = 0.0f;
      for (int l = 0; l < kL; ++l) { float e = fast_exp(row[l] - mx); row[l] = e; s += e; }
      float inv = fast_rcp(s);
      for (int l = 0; l < kL; ++l) row[l] *= inv;
    }
    __syncthreads();
    // ---- P2c: contexts c = sum_l alpha_l * enc_l (shared enc reads) ----
    {
      int m = tid >> 4, seg = tid & 15;
      const __bf16* eb = enc_bf + (size_t)(b0 + m) * kL * kH;
      for (int cc = 0; cc < 32; cc += 8) {
        int c0 = seg * 32 + cc;
        float a1c[8] = {}, a2c[8] = {};
        for (int l = 0; l < kL; ++l) {
          float a1 = logit[m * kL + l];
          float a2 = logit[(16 + m) * kL + l];
          bf16x8 ev = *(const bf16x8*)(eb + (size_t)l * kH + c0);
#pragma unroll
          for (int u = 0; u < 8; ++u) {
            float e = (float)ev[u];
            a1c[u] = fmaf(a1, e, a1c[u]);
            a2c[u] = fmaf(a2, e, a2c[u]);
          }
        }
#pragma unroll
        for (int u = 0; u < 8; ++u) {
          int fi = afrag_idx(m, c0 + u);
          cfrag[fi] = (__bf16)a1c[u];
          cfrag[kTileElems + fi] = (__bf16)a2c[u];
        }
      }
    }
    __syncthreads();
    // ---- P3: dual 2-layer GRU, fused gates + state update ----
    for (int dec = 0; dec < 2; ++dec) {
      const float* bihL = dec ? g2bih : g1bih;
      const float* bhhL = dec ? g2bhh : g1bhh;
      for (int layer = 0; layer < 2; ++layer) {
        const __bf16* xf = (layer == 0)
            ? (cfrag + (size_t)dec * kTileElems)
            : (hfrag + (size_t)(nxt * 4 + dec * 2) * kTileElems);      // layer0's new h
        const __bf16* hfr = hfrag + (size_t)(cur * 4 + dec * 2 + layer) * kTileElems;
        __bf16* hw = hfrag + (size_t)(nxt * 4 + dec * 2 + layer) * kTileElems;
        const __bf16* wih = wgru + (size_t)((dec * 2 + layer) * 2 + 0) * ((size_t)kH * k3H);
        const __bf16* whh = wgru + (size_t)((dec * 2 + layer) * 2 + 1) * ((size_t)kH * k3H);
        const float* bi = bihL + layer * k3H;
        const float* bh = bhhL + layer * k3H;
        float pvec[8] = {};
#pragma unroll 1
        for (int q = 0; q < 4; ++q) {
          int ntc = wv * 4 + q;
          int colr = ntc * 16 + nn16;
          v8f accr = {}, accz = {}, accni = {}, accnh = {};

          // base+immediate addressing; per-kt strides fit the 24-bit IOFFSET
          const int laneE = lane << 4;
          const __bf16* xp = xf + laneE;
          const __bf16* hp = hfr + laneE;
          const __bf16* bwp = wih + ((size_t)ntc << 9) + laneE;
          const __bf16* bhp = whh + ((size_t)ntc << 9) + laneE;
          constexpr int BSTEP = 96 * 512;       // elements per kt step (NT=96)
          constexpr int GOFF = 32 * kNTStride;  // +32 N-tiles: next gate block

          // prologue: A and B fragments for kt=0
          bf16x16 ax = *(const bf16x16*)(xp);
          bf16x16 ah = *(const bf16x16*)(hp);
          bf16x16 bwr = *(const bf16x16*)(bwp);
          bf16x16 bwz = *(const bf16x16*)(bwp + GOFF);
          bf16x16 bwn = *(const bf16x16*)(bwp + 2 * GOFF);
          bf16x16 bhr = *(const bf16x16*)(bhp);
          bf16x16 bhz = *(const bf16x16*)(bhp + GOFF);
          bf16x16 bhn = *(const bf16x16*)(bhp + 2 * GOFF);
          __builtin_amdgcn_sched_group_barrier(SG_DS_RD, 4, 0);
          __builtin_amdgcn_sched_group_barrier(SG_VMEM_RD, 12, 0);
#pragma unroll
          for (int kt = 0; kt < kKIters; ++kt) {
            bf16x16 axn, ahn, nwr, nwz, nwn, nhr, nhz, nhn;
            if (kt < kKIters - 1) {            // prefetch next K-step's fragments
              const __bf16* bq = bwp + (size_t)(kt + 1) * BSTEP;
              const __bf16* cq = bhp + (size_t)(kt + 1) * BSTEP;
              axn = *(const bf16x16*)(xp + (kt + 1) * 512);
              ahn = *(const bf16x16*)(hp + (kt + 1) * 512);
              nwr = *(const bf16x16*)(bq);
              nwz = *(const bf16x16*)(bq + GOFF);
              nwn = *(const bf16x16*)(bq + 2 * GOFF);
              nhr = *(const bf16x16*)(cq);
              nhz = *(const bf16x16*)(cq + GOFF);
              nhn = *(const bf16x16*)(cq + 2 * GOFF);
            }
            accr  = WMMA_BF16(ax, bwr, accr);
            accz  = WMMA_BF16(ax, bwz, accz);
            accni = WMMA_BF16(ax, bwn, accni);
            accr  = WMMA_BF16(ah, bhr, accr);
            accz  = WMMA_BF16(ah, bhz, accz);
            accnh = WMMA_BF16(ah, bhn, accnh);
            __builtin_amdgcn_sched_group_barrier(SG_DS_RD, 4, 0);
            __builtin_amdgcn_sched_group_barrier(SG_VMEM_RD, 12, 0);
            __builtin_amdgcn_sched_group_barrier(SG_WMMA, 6, 0);
            if (kt < kKIters - 1) {
              ax = axn; ah = ahn;
              bwr = nwr; bwz = nwz; bwn = nwn;
              bhr = nhr; bhz = nhz; bhn = nhn;
            }
          }
          // fused activation + state update (lane holds same (row,col) in all 4 accs)
          float br = bi[colr] + bh[colr];
          float bz = bi[kH + colr] + bh[kH + colr];
          float bni = bi[2 * kH + colr];
          float bnh = bh[2 * kH + colr];
          float wo = (layer == 1) ? out_w[dec * kH + colr] : 0.0f;
          int fi0 = afrag_idx(0, colr);
#pragma unroll
          for (int i = 0; i < 8; ++i) {
            int row = half * 8 + i;
            int fi = fi0 + (row << 4);
            float rr = fast_sigmoid(accr[i] + br);
            float zz = fast_sigmoid(accz[i] + bz);
            float nv = fast_tanh((accni[i] + bni) + rr * (accnh[i] + bnh));
            float hold = (float)hfr[fi];
            float hnew = (1.0f - zz) * nv + zz * hold;
            hw[fi] = (__bf16)hnew;
            pvec[i] = fmaf(hnew, wo, pvec[i]);
          }
        }
        if (layer == 1) {
          // reduce pvec over the 16 lanes of each half-group (rows live per half)
#pragma unroll
          for (int msk = 1; msk < 16; msk <<= 1) {
#pragma unroll
            for (int i = 0; i < 8; ++i) pvec[i] += __shfl_xor(pvec[i], msk, 32);
          }
          if (nn16 == 0) {
#pragma unroll
            for (int i = 0; i < 8; ++i) red2[wv * 16 + half * 8 + i] = pvec[i];
          }
        }
        __syncthreads();
        if (layer == 1 && tid < 16) {
          float s = 0.0f;
#pragma unroll
          for (int w = 0; w < 8; ++w) s += red2[w * 16 + tid];
          float* dp = dec ? d2 : d1;
          dp[(size_t)(b0 + tid) * kL + step] = s;
        }
      }
    }
    cur = nxt;
  }
}

// final = sigmoid(tanh(d1[b,t] + d2[b, idx(t)] + out_b))
__global__ void out_kernel(const float* __restrict__ d1, const float* __restrict__ d2,
                           const float* __restrict__ out_b, float* __restrict__ out) {
  int idx = blockIdx.x * blockDim.x + threadIdx.x;
  int b = idx >> 6, t = idx & 63;
  int tj = (t >= kL - kD - 1) ? (kL - 1) : (t + kD);
  float v = d1[idx] + d2[b * kL + tj] + out_b[0];
  out[idx] = fast_sigmoid(fast_tanh(v));
}

extern "C" void kernel_launch(void* const* d_in, const int* in_sizes, int n_in,
                              void* d_out, int out_size, void* d_ws, size_t ws_size,
                              hipStream_t stream) {
  (void)in_sizes; (void)n_in; (void)out_size; (void)ws_size;
  const float* received = (const float*)d_in[0];
  const float* fc_w   = (const float*)d_in[1];
  const float* fc_b   = (const float*)d_in[2];
  const float* attn_w = (const float*)d_in[3];
  const float* v_w    = (const float*)d_in[4];
  const float* g1wih  = (const float*)d_in[5];
  const float* g1whh  = (const float*)d_in[6];
  const float* g1bih  = (const float*)d_in[7];
  const float* g1bhh  = (const float*)d_in[8];
  const float* g2wih  = (const float*)d_in[9];
  const float* g2whh  = (const float*)d_in[10];
  const float* g2bih  = (const float*)d_in[11];
  const float* g2bhh  = (const float*)d_in[12];
  const float* out_w  = (const float*)d_in[13];
  const float* out_b  = (const float*)d_in[14];

  char* ws = (char*)d_ws;
  size_t off = 0;
  __bf16* enc_bf  = (__bf16*)(ws + off); off += (size_t)kB * kL * kH * 2;
  __bf16* encproj = (__bf16*)(ws + off); off += (size_t)kB * kL * kH * 2;
  __bf16* w1t     = (__bf16*)(ws + off); off += (size_t)kH * kH * 2;
  __bf16* w2t     = (__bf16*)(ws + off); off += (size_t)kH * kH * 2;
  __bf16* wgru    = (__bf16*)(ws + off); off += (size_t)8 * kH * k3H * 2;
  float*  d1      = (float*)(ws + off);  off += (size_t)kB * kL * 4;
  float*  d2      = (float*)(ws + off);  off += (size_t)kB * kL * 4;

  // Weights -> bf16 WMMA B-fragment layout (L2-resident thereafter).
  pack_bfrag_kernel<<<1024, 256, 0, stream>>>(attn_w, w1t, kH, 2 * kH, 0);
  pack_bfrag_kernel<<<1024, 256, 0, stream>>>(attn_w, w2t, kH, 2 * kH, kH);
  for (int dec = 0; dec < 2; ++dec) {
    const float* wihb = dec ? g2wih : g1wih;
    const float* whhb = dec ? g2whh : g1whh;
    for (int layer = 0; layer < 2; ++layer) {
      size_t slot = (size_t)((dec * 2 + layer) * 2);
      pack_bfrag_kernel<<<3072, 256, 0, stream>>>(
          wihb + (size_t)layer * k3H * kH, wgru + slot * (size_t)kH * k3H, k3H, kH, 0);
      pack_bfrag_kernel<<<3072, 256, 0, stream>>>(
          whhb + (size_t)layer * k3H * kH, wgru + (slot + 1) * (size_t)kH * k3H, k3H, kH, 0);
    }
  }
  encode_kernel<<<(kB * kL * kH) / 256, 256, 0, stream>>>(received, fc_w, fc_b, enc_bf);
  encproj_kernel<<<(kB * kL) / 16, 256, 0, stream>>>(enc_bf, w2t, encproj);

  constexpr size_t shmem = (size_t)8 * kTileElems * 2 +   // h frags, double-buffered
                           (size_t)2 * kTileElems * 2 +   // context frags
                           (size_t)2 * kTileElems * 4 +   // sproj
                           (size_t)2 * 16 * kL * 4 +      // logits
                           128 * 4 + (size_t)kH * 4;      // red2 + v_w  (= 240128 B)
  (void)hipFuncSetAttribute(reinterpret_cast<const void*>(decoder_kernel),
                            hipFuncAttributeMaxDynamicSharedMemorySize, (int)shmem);
  decoder_kernel<<<kB / 16, 256, shmem, stream>>>(enc_bf, encproj, w1t, wgru,
                                                  g1bih, g1bhh, g2bih, g2bhh,
                                                  v_w, out_w, d1, d2);
  out_kernel<<<(kB * kL) / 256, 256, 0, stream>>>(d1, d2, out_b, (float*)d_out);
}
